// Controller_73229192397155
// MI455X (gfx1250) — matrix operand order repaired and checked by reading, not compiled, not run on hardware
//
#include <hip/hip_runtime.h>
#include <cstddef>

#define M 2048
#define R 16
#define NINS 6
#define T_MAX 32
#define MULT 5.0f
#define STOP_THRESHOLD 0.9f

typedef __attribute__((ext_vector_type(16))) _Float16 v16h;
typedef __attribute__((ext_vector_type(8)))  float    v8f;

// ---------------- block reduction helpers ----------------
__device__ __forceinline__ float blockReduceSum(float v, float* sdata) {
  int t = threadIdx.x;
  sdata[t] = v; __syncthreads();
  for (int s = blockDim.x >> 1; s > 0; s >>= 1) {
    if (t < s) sdata[t] += sdata[t + s];
    __syncthreads();
  }
  float r = sdata[0]; __syncthreads();
  return r;
}
__device__ __forceinline__ float blockReduceMax(float v, float* sdata) {
  int t = threadIdx.x;
  sdata[t] = v; __syncthreads();
  for (int s = blockDim.x >> 1; s > 0; s >>= 1) {
    if (t < s) sdata[t] = fmaxf(sdata[t], sdata[t + s]);
    __syncthreads();
  }
  float r = sdata[0]; __syncthreads();
  return r;
}

// ---------------- init: regs = softmax(MULT*regs0, rows), ir = softmax(MULT*ir0) -----
__global__ void init_kernel(const float* __restrict__ regs0, const float* __restrict__ ir0,
                            float* __restrict__ regs, float* __restrict__ ir,
                            float* __restrict__ scal, int* __restrict__ flag) {
  __shared__ float red[256];
  int b = blockIdx.x, t = threadIdx.x;
  if (b < R) {
    const float* src = regs0 + (size_t)b * M;
    float mx = -3.0e38f;
    for (int j = t; j < M; j += 256) mx = fmaxf(mx, MULT * src[j]);
    mx = blockReduceMax(mx, red);
    float s = 0.f;
    for (int j = t; j < M; j += 256) s += __expf(MULT * src[j] - mx);
    s = blockReduceSum(s, red);
    float inv = 1.f / s;
    for (int j = t; j < M; j += 256) regs[(size_t)b * M + j] = __expf(MULT * src[j] - mx) * inv;
  } else { // b == R: ir softmax + scalar state
    float mx = -3.0e38f;
    for (int j = t; j < M; j += 256) mx = fmaxf(mx, MULT * ir0[j]);
    mx = blockReduceMax(mx, red);
    float s = 0.f;
    for (int j = t; j < M; j += 256) s += __expf(MULT * ir0[j] - mx);
    s = blockReduceSum(s, red);
    float inv = 1.f / s;
    for (int j = t; j < M; j += 256) ir[j] = __expf(MULT * ir0[j] - mx) * inv;
    if (t == 0) { scal[0] = 0.f; /*sp*/ scal[1] = 0.f; /*conf*/ scal[2] = 1.f; /*eff*/ *flag = 1; }
  }
}

// One controller head: logits = MULT * (src @ ir), softmax into shared + global.
// (Plain pointer params -> runtime addrspacecast, avoids illegal static LDS-ptr init.)
__device__ void head_softmax(const float* __restrict__ src, int count,
                             const float* __restrict__ ir,
                             float* sOut, float* __restrict__ gOut, float* red) {
  int t = threadIdx.x;
  for (int r = 0; r < count; ++r) {
    float loc = 0.f;
    const float* row = src + (size_t)r * M;
    for (int j = t; j < M; j += 256) loc += row[j] * ir[j];
    float s = blockReduceSum(loc, red);
    if (t == 0) sOut[r] = MULT * s;
  }
  __syncthreads();
  if (t == 0) {
    float mx = sOut[0];
    for (int r = 1; r < count; ++r) mx = fmaxf(mx, sOut[r]);
    float s = 0.f;
    for (int r = 0; r < count; ++r) { sOut[r] = __expf(sOut[r] - mx); s += sOut[r]; }
    float inv = 1.f / s;
    for (int r = 0; r < count; ++r) { sOut[r] *= inv; gOut[r] = sOut[r]; }
  }
  __syncthreads();
}

// ---------------- per-step controller: a,b,o,e softmaxes; va,vb,w ----------------
__global__ void controller_kernel(const float* __restrict__ fa0, const float* __restrict__ sa0,
                                  const float* __restrict__ op0, const float* __restrict__ ins0,
                                  const float* __restrict__ regs, const float* __restrict__ ir,
                                  float* __restrict__ aW, float* __restrict__ bW,
                                  float* __restrict__ oW, float* __restrict__ eW,
                                  float* __restrict__ va, float* __restrict__ vb,
                                  float* __restrict__ wv, const int* __restrict__ flag) {
  if (!*flag) return;
  __shared__ float red[256];
  __shared__ float sA[R], sB[R], sO[R], sE[NINS];
  int t = threadIdx.x;

  head_softmax(fa0,  R,    ir, sA, aW, red);
  head_softmax(sa0,  R,    ir, sB, bW, red);
  head_softmax(op0,  R,    ir, sO, oW, red);
  head_softmax(ins0, NINS, ir, sE, eW, red);

  // va = a @ regs ; vb = b @ regs ; w = e5 * va
  float e5 = sE[5];
  for (int j = t; j < M; j += 256) {
    float sva = 0.f, svb = 0.f;
    for (int r = 0; r < R; ++r) {
      float rv = regs[(size_t)r * M + j];
      sva = fmaf(sA[r], rv, sva);
      svb = fmaf(sB[r], rv, svb);
    }
    va[j] = sva; vb[j] = svb; wv[j] = e5 * sva;
  }
}

// ---------------- circular conv (add) and correlation (sub), direct O(M^2) ------------
__global__ void conv_kernel(const float* __restrict__ va, const float* __restrict__ vb,
                            float* __restrict__ addv, float* __restrict__ subv,
                            const int* __restrict__ flag) {
  if (!*flag) return;
  __shared__ float sva[M], svb[M];
  int t = threadIdx.x;
  for (int j = t; j < M; j += blockDim.x) { sva[j] = va[j]; svb[j] = vb[j]; }
  __syncthreads();
  int k = blockIdx.x * blockDim.x + t;
  float sadd = 0.f, ssub = 0.f;
  for (int i = 0; i < M; ++i) {
    float x = sva[i];
    sadd = fmaf(x, svb[(k - i) & (M - 1)], sadd);
    ssub = fmaf(x, svb[(i - k) & (M - 1)], ssub);
  }
  addv[k] = sadd; subv[k] = ssub;
}

// ---------------- fused mem pass: WMMA vread partials + rank-1 update + corr ----------
// One wave per block; tile = 32 rows x 256 cols of mem, staged in LDS.
// vread contraction (K=32) via v_wmma_f32_16x16x32_f16 with hi/lo fp16 split of va
// (rows 0/1 of the A matrix) for near-fp32 accuracy at zero bandwidth cost.
#define TILE_R 32
#define TILE_C 256
__global__ void __launch_bounds__(32)
mem_pass_kernel(float* __restrict__ mem, const float* __restrict__ outmem,
                const float* __restrict__ mask, const float* __restrict__ va,
                const float* __restrict__ vb, const float* __restrict__ wv,
                float* __restrict__ vreadPart, float* __restrict__ corrPart,
                const int* __restrict__ flag) {
  if (!*flag) return;
  __shared__ float tile[TILE_R][TILE_C + 5];
  __shared__ float svb[TILE_C];
  __shared__ _Float16 vhi[TILE_R], vlo[TILE_R];
  __shared__ float sw[TILE_R];

  const int lane = threadIdx.x;
  const int rowBase = blockIdx.y * TILE_R;
  const int colBase = blockIdx.x * TILE_C;

  {
    float vaf = va[rowBase + lane];
    _Float16 h = (_Float16)vaf;
    vhi[lane] = h;
    vlo[lane] = (_Float16)(vaf - (float)h);
    sw[lane] = wv[rowBase + lane];
  }
  for (int j = lane; j < TILE_C; j += 32) svb[j] = vb[colBase + j];
  for (int r = 0; r < TILE_R; ++r) {
    size_t base = (size_t)(rowBase + r) * M + colBase;
    __builtin_prefetch(&outmem[base], 0, 0);
    __builtin_prefetch(&mask[base], 0, 0);
    for (int j = lane; j < TILE_C; j += 32) tile[r][j] = mem[base + j];
  }
  __syncthreads();

  // ---- A matrix pack: 16x32 f16. Row 0 = va_hi chunk, row 1 = va_lo chunk, rest 0.
  // Doc layout (ISA 7.12.2): lanes 0-15 row M=lane hold K={0..7,16..23};
  // lanes 16-31 hold K={8..15,24..31}.
  v16h A;
  for (int h = 0; h < 16; ++h) A[h] = (_Float16)0.0f;
  {
    int mrow = lane & 15, sel = lane >> 4;
    if (mrow < 2) {
      const _Float16* src = (mrow == 0) ? (const _Float16*)vhi : (const _Float16*)vlo;
      for (int h = 0; h < 8; ++h) {
        A[h]     = src[sel * 8 + h];
        A[8 + h] = src[16 + sel * 8 + h];
      }
    }
  }
  v8f cz;
  for (int i = 0; i < 8; ++i) cz[i] = 0.f;

  // ---- 16 column tiles of 16: D(16x16) = A(16x32) x B(32x16), fp32 accum.
  // B layout: lane n (0..15) holds column n, K={0..7,16..23} in its 16 halves;
  // lane n+16 holds column n, K={8..15,24..31} (matches A).
  const int col = lane & 15, kb = (lane >> 4) * 8;
  for (int t = 0; t < 16; ++t) {
    int j0 = t * 16;
    v16h Bh, Bl;
    for (int h = 0; h < 16; ++h) {
      float v = tile[kb + h + (h & 8)][j0 + col];
      _Float16 hh = (_Float16)v;
      Bh[h] = hh;
      Bl[h] = (_Float16)(v - (float)hh);
    }
    v8f c1 = __builtin_amdgcn_wmma_f32_16x16x32_f16(false, A, false, Bh, (short)0, cz, false, false);
    v8f c2 = __builtin_amdgcn_wmma_f32_16x16x32_f16(false, A, false, Bl, (short)0, cz, false, false);
    // D rows 0/1 live in VGPRs 0/1 of lanes 0-15 (ISA C/D layout).
    float p = c1[0] + c1[1] + c2[0];
    if (lane < 16) vreadPart[(size_t)blockIdx.y * M + colBase + j0 + lane] = p;
  }

  // ---- rank-1 memory update + masked squared-error accumulation (new mem) ----
  float acc = 0.f;
  for (int r = 0; r < TILE_R; ++r) {
    float wr = sw[r];
    size_t base = (size_t)(rowBase + r) * M + colBase;
    for (int j = lane; j < TILE_C; j += 32) {
      float old = tile[r][j];
      float nv = old * (1.f - wr) + wr * svb[j];
      mem[base + j] = nv;
      float d = outmem[base + j] - nv;
      acc = fmaf(mask[base + j], d * d, acc);
    }
  }
  for (int off = 16; off > 0; off >>= 1) acc += __shfl_down(acc, off, 32);
  if (lane == 0) corrPart[blockIdx.y * gridDim.x + blockIdx.x] = acc;
}

// ---------------- per-step finalize: vread reduce, regs update, ir roll, scalars ------
__global__ void finalize_kernel(float* __restrict__ regs, float* __restrict__ ir,
                                const float* __restrict__ va, const float* __restrict__ addv,
                                const float* __restrict__ subv, const float* __restrict__ vreadPart,
                                float* __restrict__ vread, const float* __restrict__ corrPart,
                                const float* __restrict__ oW, const float* __restrict__ eW,
                                float* __restrict__ scal, int* __restrict__ flag) {
  if (!*flag) return;
  __shared__ float sIr[M];
  __shared__ float red[256];
  int t = threadIdx.x;

  // deterministic reduction of vread partials over the 64 row groups
  for (int j = t; j < M; j += 256) {
    float s = 0.f;
    for (int g = 0; g < 64; ++g) s += vreadPart[(size_t)g * M + j];
    vread[j] = s;
  }
  float e0 = eW[0], e1 = eW[1], e2 = eW[2], e3 = eW[3], e4 = eW[4], e5 = eW[5];

  // regs = regs*(1-o) + o*out_val   (each thread owns its columns j)
  for (int j = t; j < M; j += 256) {
    float ov = (e0 + e5) * va[j] + e2 * addv[j] + e3 * subv[j] + e4 * vread[j];
    if (j == 0) ov += e1; // "zero" one-hot vector at index 0
    for (int r = 0; r < R; ++r) {
      float o = oW[r];
      size_t idx = (size_t)r * M + j;
      regs[idx] = regs[idx] * (1.f - o) + o * ov;
    }
  }

  // ir = roll(ir, 1)
  for (int j = t; j < M; j += 256) sIr[j] = ir[j];
  __syncthreads();
  for (int j = t; j < M; j += 256) ir[j] = sIr[(j + M - 1) & (M - 1)];

  // corr_t reduction (512 block partials) + scalar state update
  float c = corrPart[t] + corrPart[t + 256];
  c = blockReduceSum(c, red);
  if (t == 0) {
    float sp = scal[0];
    float nsp = sp + e0 * (1.f - sp);
    scal[1] += (nsp - sp) * c;   // conf
    scal[2] += 1.f - nsp;        // eff
    scal[0] = nsp;
    *flag = (nsp < STOP_THRESHOLD) ? 1 : 0;
  }
}

// ---------------- final correctness reduction + weighted output ----------------------
__global__ void corr_final_kernel(const float* __restrict__ mem, const float* __restrict__ outmem,
                                  const float* __restrict__ mask, float* __restrict__ part) {
  __shared__ float red[256];
  size_t idx = (size_t)blockIdx.x * blockDim.x + threadIdx.x;
  size_t stride = (size_t)gridDim.x * blockDim.x;
  float acc = 0.f;
  for (size_t k = idx; k < (size_t)M * M; k += stride) {
    float d = outmem[k] - mem[k];
    acc = fmaf(mask[k], d * d, acc);
  }
  acc = blockReduceSum(acc, red);
  if (threadIdx.x == 0) part[blockIdx.x] = acc;
}

__global__ void output_kernel(const float* __restrict__ part, const float* __restrict__ scal,
                              float* __restrict__ out) {
  __shared__ float red[256];
  float acc = part[threadIdx.x] + part[threadIdx.x + 256];
  acc = blockReduceSum(acc, red);
  if (threadIdx.x == 0) {
    float correctness = acc;
    float sp = scal[0], conf = scal[1], eff = scal[2];
    float halting = (sp < STOP_THRESHOLD) ? (1.f - sp) : 0.f;
    out[0] = 0.2f * correctness + 0.2f * conf + 0.2f * halting + 0.4f * eff;
  }
}

// ---------------- host-side orchestration (graph-capture safe) -----------------------
extern "C" void kernel_launch(void* const* d_in, const int* in_sizes, int n_in,
                              void* d_out, int out_size, void* d_ws, size_t ws_size,
                              hipStream_t stream) {
  (void)in_sizes; (void)n_in; (void)out_size; (void)ws_size;
  const float* memory  = (const float*)d_in[0];
  const float* outmem  = (const float*)d_in[1];
  const float* mask    = (const float*)d_in[2];
  const float* fa0     = (const float*)d_in[3];
  const float* sa0     = (const float*)d_in[4];
  const float* op0     = (const float*)d_in[5];
  const float* ins0    = (const float*)d_in[6];
  const float* regs0   = (const float*)d_in[7];
  const float* ir0     = (const float*)d_in[8];

  float* ws = (float*)d_ws;
  size_t off = 0;
  float* mem       = ws + off; off += (size_t)M * M;   // 16 MB evolving memory
  float* regs      = ws + off; off += (size_t)R * M;
  float* ir        = ws + off; off += M;
  float* va        = ws + off; off += M;
  float* vb        = ws + off; off += M;
  float* wv        = ws + off; off += M;
  float* vread     = ws + off; off += M;
  float* addv      = ws + off; off += M;
  float* subv      = ws + off; off += M;
  float* vreadPart = ws + off; off += (size_t)64 * M;  // per-rowgroup GEMV partials
  float* corrPart  = ws + off; off += 512;
  float* corrFin   = ws + off; off += 512;
  float* aW        = ws + off; off += 16;
  float* bW        = ws + off; off += 16;
  float* oW        = ws + off; off += 16;
  float* eW        = ws + off; off += 8;
  float* scal      = ws + off; off += 4;               // sp, conf, eff
  int*   flag      = (int*)(ws + off); off += 4;

  (void)hipMemcpyAsync(mem, memory, (size_t)M * M * sizeof(float),
                       hipMemcpyDeviceToDevice, stream);
  init_kernel<<<R + 1, 256, 0, stream>>>(regs0, ir0, regs, ir, scal, flag);

  for (int t = 0; t < T_MAX; ++t) {
    controller_kernel<<<1, 256, 0, stream>>>(fa0, sa0, op0, ins0, regs, ir,
                                             aW, bW, oW, eW, va, vb, wv, flag);
    conv_kernel<<<16, 128, 0, stream>>>(va, vb, addv, subv, flag);
    mem_pass_kernel<<<dim3(M / TILE_C, M / TILE_R), 32, 0, stream>>>(
        mem, outmem, mask, va, vb, wv, vreadPart, corrPart, flag);
    finalize_kernel<<<1, 256, 0, stream>>>(regs, ir, va, addv, subv, vreadPart,
                                           vread, corrPart, oW, eW, scal, flag);
  }
  corr_final_kernel<<<512, 256, 0, stream>>>(mem, outmem, mask, corrFin);
  output_kernel<<<1, 256, 0, stream>>>(corrFin, scal, (float*)d_out);
}
